// Quantizer_10350871183376
// MI455X (gfx1250) — compile-verified, matrix operand order checked by reading, not run on hardware
//
#include <hip/hip_runtime.h>
#include <hip/hip_bf16.h>
#include <cstdint>

typedef __bf16 bf16_t;
typedef __attribute__((ext_vector_type(16))) __bf16 v16bf;
typedef __attribute__((ext_vector_type(8)))  __bf16 v8bf;
typedef __attribute__((ext_vector_type(8)))  float  v8f;

#define D_DIM 256
#define K_DIM 1024
#define B_DIM 65536
#define ROWS_PER_BLOCK 128
#define WAVES_PER_BLOCK 8
#define NUM_BLOCKS (B_DIM / ROWS_PER_BLOCK)   // 512
#define LDS_ROW 264                            // 256 bf16 + 8 bf16 pad (528 B rows)

// ---------------------------------------------------------------------------
// CDNA5 async global->LDS copy (ASYNCcnt path). Inline asm is portable across
// both gfx1250 toolchains; LDS dest offset rides in a VGPR, 64-bit global
// address in a VGPR pair.
// ---------------------------------------------------------------------------
__device__ __forceinline__ void async_copy_b128_to_lds(uint32_t lds_off, uint64_t gaddr) {
  asm volatile("global_load_async_to_lds_b128 %0, %1, off"
               :: "v"(lds_off), "v"(gaddr)
               : "memory");
}
__device__ __forceinline__ void wait_asynccnt0() {
  asm volatile("s_wait_asynccnt 0x0" ::: "memory");
}

// ---------------------------------------------------------------------------
// Kernel 1: codebook prep. W (f32, K x D) -> Wb (bf16, K x D) and w2[k]=||w_k||^2.
// Wb is 512 KB -> resident in the 192 MB L2 for the whole main kernel.
// ---------------------------------------------------------------------------
__global__ __launch_bounds__(256) void vq_prep_kernel(
    const float* __restrict__ W, bf16_t* __restrict__ Wb, float* __restrict__ w2) {
  const int k = blockIdx.x;    // 0..1023
  const int d = threadIdx.x;   // 0..255
  const float w = W[(size_t)k * D_DIM + d];
  Wb[(size_t)k * D_DIM + d] = (bf16_t)w;

  __shared__ float red[256];
  red[d] = w * w;
  __syncthreads();
  #pragma unroll
  for (int s = 128; s > 0; s >>= 1) {
    if (d < s) red[d] += red[d + s];
    __syncthreads();
  }
  if (d == 0) w2[k] = red[0];
}

// ---------------------------------------------------------------------------
// Kernel 2: main VQ kernel. Each wave owns 16 rows of x (A-fragments resident
// in VGPRs). B-tiles (16 codes x 256 d, bf16) are double-buffered into LDS via
// global_load_async_to_lds_b128 (one copy per block instead of one per wave:
// 8x less L2 traffic), consumed with ds_load_b128, and fed to
// v_wmma_f32_16x16x32_bf16 with f32 accumulation.
// ---------------------------------------------------------------------------
__global__ __launch_bounds__(256) void vq_main_kernel(
    const float*  __restrict__ x,        // [B, D] f32
    const float*  __restrict__ W,        // [K, D] f32 (for exact gather)
    const bf16_t* __restrict__ Wb,       // [K, D] bf16
    const float*  __restrict__ w2,       // [K]
    float* __restrict__ outQuant,        // [B, D]
    float* __restrict__ outIdx,          // [B] (indices stored as float values)
    float* __restrict__ partials) {      // [NUM_BLOCKS] loss partial sums

  __shared__ __align__(16) bf16_t bufB[2][16][LDS_ROW]; // 2 x 16 x 528B = 16.5 KB
  __shared__ float sval[WAVES_PER_BLOCK][16][16];
  __shared__ int   sidx[WAVES_PER_BLOCK][16][16];
  __shared__ int   rowk[ROWS_PER_BLOCK];
  __shared__ float lred[256];

  const int tid  = threadIdx.x;
  const int wave = tid >> 5;
  const int lane = tid & 31;
  const int l16  = lane & 15;
  const int h    = lane >> 4;          // half-wave select (ISA K-chunk split)

  const int blockRow = blockIdx.x * ROWS_PER_BLOCK;
  const int waveRow  = blockRow + wave * 16;

  // Stage B-tile t (16 codes x 512 B) into bufB[pb]: 512 x 16B chunks over
  // 256 threads -> 2 async b128 copies per thread.
  auto stage_tile = [&](int t, int pb) {
    const int n0 = t * 16;
    #pragma unroll
    for (int i = 0; i < 2; ++i) {
      const int c   = tid + i * 256;   // 0..511
      const int row = c >> 5;          // code within tile
      const int col = c & 31;          // 16B chunk within 512B row
      const uint32_t lds_off =
          (uint32_t)(uintptr_t)&bufB[pb][row][col * 8];
      const bf16_t* g = Wb + (size_t)(n0 + row) * D_DIM + col * 8;
      async_copy_b128_to_lds(lds_off, (uint64_t)(uintptr_t)g);
    }
  };

  // ---- Load this wave's 16x256 x-tile once; convert to 8 bf16 A-fragments.
  // 16-bit A layout (16x32): lanes 0-15 hold K=0..7 (v0-3) and K=16..23 (v4-7);
  // lanes 16-31 hold K=8..15 and K=24..31.
  v16bf a[8];
  const float* xrow = x + (size_t)(waveRow + l16) * D_DIM;
  #pragma unroll
  for (int s = 0; s < 8; ++s) {
    const float4* p0 = (const float4*)(xrow + s * 32 + h * 8);
    const float4* p1 = (const float4*)(xrow + s * 32 + 16 + h * 8);
    float4 f0 = p0[0], f1 = p0[1];
    float4 f2 = p1[0], f3 = p1[1];
    v16bf av;
    av[0]  = (bf16_t)f0.x; av[1]  = (bf16_t)f0.y; av[2]  = (bf16_t)f0.z; av[3]  = (bf16_t)f0.w;
    av[4]  = (bf16_t)f1.x; av[5]  = (bf16_t)f1.y; av[6]  = (bf16_t)f1.z; av[7]  = (bf16_t)f1.w;
    av[8]  = (bf16_t)f2.x; av[9]  = (bf16_t)f2.y; av[10] = (bf16_t)f2.z; av[11] = (bf16_t)f2.w;
    av[12] = (bf16_t)f3.x; av[13] = (bf16_t)f3.y; av[14] = (bf16_t)f3.z; av[15] = (bf16_t)f3.w;
    a[s] = av;
  }

  float minv[8];
  int   mini[8];
  #pragma unroll
  for (int r = 0; r < 8; ++r) { minv[r] = 3.402823466e+38f; mini[r] = 0; }

  // ---- Prologue: stage tile 0.
  stage_tile(0, 0);
  wait_asynccnt0();
  __syncthreads();

  // ---- Sweep all 1024 codes: 64 tiles of 16 codes, double-buffered in LDS.
  for (int t = 0; t < 64; ++t) {
    const int pb = t & 1;
    if (t + 1 < 64) stage_tile(t + 1, pb ^ 1);   // prefetch next tile (async)

    const int n0 = t * 16;
    const bf16_t* wrow = &bufB[pb][l16][0];      // column n0+l16 of B = code row
    v8f c = {};
    #pragma unroll
    for (int s = 0; s < 8; ++s) {
      v8bf b0 = *(const v8bf*)(wrow + s * 32 + h * 8);        // ds_load_b128
      v8bf b1 = *(const v8bf*)(wrow + s * 32 + 16 + h * 8);   // ds_load_b128
      v16bf bv;
      #pragma unroll
      for (int i = 0; i < 8; ++i) { bv[i] = b0[i]; bv[8 + i] = b1[i]; }
      c = __builtin_amdgcn_wmma_f32_16x16x32_bf16(
          /*neg_a=*/false, a[s], /*neg_b=*/false, bv,
          /*c_mod=*/(short)0, c, /*reuse_a=*/false, /*reuse_b=*/false);
    }

    // dist2(m,n) = x2(m) + w2(n) - 2*dot ; x2 constant per row -> dropped.
    const float w2n = w2[n0 + l16];
    const int   nId = n0 + l16;
    #pragma unroll
    for (int r = 0; r < 8; ++r) {
      const float dv = w2n - 2.0f * c[r];
      if (dv < minv[r]) { minv[r] = dv; mini[r] = nId; }  // ascending n -> first min kept
    }

    wait_asynccnt0();    // my async copies for tile t+1 have landed in LDS
    __syncthreads();     // everyone's copies landed; tile t fully consumed
  }

  // ---- Cross-lane argmin: C layout is row (r + 8*h) x column l16.
  #pragma unroll
  for (int r = 0; r < 8; ++r) {
    sval[wave][r + 8 * h][l16] = minv[r];
    sidx[wave][r + 8 * h][l16] = mini[r];
  }
  __syncthreads();

  if (tid < ROWS_PER_BLOCK) {
    const int row = tid;
    const int wv = row >> 4, rl = row & 15;
    float bv = sval[wv][rl][0];
    int   bi = sidx[wv][rl][0];
    #pragma unroll
    for (int c2 = 1; c2 < 16; ++c2) {
      const float v  = sval[wv][rl][c2];
      const int   i2 = sidx[wv][rl][c2];
      if (v < bv || (v == bv && i2 < bi)) { bv = v; bi = i2; }  // tie -> lowest index
    }
    rowk[row] = bi;
    outIdx[blockRow + row] = (float)bi;
  }
  __syncthreads();

  // ---- Phase 2: exact f32 gather of W[idx], write quant_out (coalesced:
  // at step j all 256 lanes cover one full row), accumulate loss partial.
  float lacc = 0.0f;
  for (int e = tid; e < ROWS_PER_BLOCK * D_DIM; e += 256) {
    const int row = e >> 8;       // D_DIM == 256
    const int d   = e & 255;
    const int k   = rowk[row];
    const float wv = W[(size_t)k * D_DIM + d];
    const float xv = x[(size_t)(blockRow + row) * D_DIM + d];
    outQuant[(size_t)(blockRow + row) * D_DIM + d] = wv;
    const float df = wv - xv;
    lacc += df * df;
  }
  lred[tid] = lacc;
  __syncthreads();
  #pragma unroll
  for (int s = 128; s > 0; s >>= 1) {
    if (tid < s) lred[tid] += lred[tid + s];
    __syncthreads();
  }
  if (tid == 0) partials[blockIdx.x] = lred[0];
}

// ---------------------------------------------------------------------------
// Kernel 3: deterministic final reduction of loss partials -> both loss slots
// (codebook_loss == commitment_loss numerically: mean((W[idx]-x)^2)).
// ---------------------------------------------------------------------------
__global__ __launch_bounds__(256) void vq_loss_kernel(
    const float* __restrict__ partials, int n, float* __restrict__ outLoss) {
  __shared__ float red[256];
  float acc = 0.0f;
  for (int i = threadIdx.x; i < n; i += 256) acc += partials[i];
  red[threadIdx.x] = acc;
  __syncthreads();
  #pragma unroll
  for (int s = 128; s > 0; s >>= 1) {
    if (threadIdx.x < s) red[threadIdx.x] += red[threadIdx.x + s];
    __syncthreads();
  }
  if (threadIdx.x == 0) {
    const float mean = red[0] / (float)((size_t)B_DIM * (size_t)D_DIM);
    outLoss[0] = mean;  // codebook_loss
    outLoss[1] = mean;  // commitment_loss
  }
}

// ---------------------------------------------------------------------------
extern "C" void kernel_launch(void* const* d_in, const int* in_sizes, int n_in,
                              void* d_out, int out_size, void* d_ws, size_t ws_size,
                              hipStream_t stream) {
  const float* x = (const float*)d_in[0];   // [65536, 256]
  const float* W = (const float*)d_in[1];   // [1024, 256]
  float* out = (float*)d_out;

  // Workspace layout
  char* ws = (char*)d_ws;
  bf16_t* Wb       = (bf16_t*)(ws);                                   // 512 KB
  float*  w2       = (float*)(ws + (size_t)K_DIM * D_DIM * 2);        // 4 KB
  float*  partials = (float*)(ws + (size_t)K_DIM * D_DIM * 2 + 4096); // 2 KB

  // Output layout: [quant_out B*D][codebook_loss][commitment_loss][idx B]
  const size_t Q = (size_t)B_DIM * D_DIM;
  float* outQuant = out;
  float* outLoss  = out + Q;
  float* outIdx   = out + Q + 2;

  vq_prep_kernel<<<K_DIM, 256, 0, stream>>>(W, Wb, w2);
  vq_main_kernel<<<NUM_BLOCKS, 256, 0, stream>>>(x, W, Wb, w2, outQuant, outIdx, partials);
  vq_loss_kernel<<<1, 256, 0, stream>>>(partials, NUM_BLOCKS, outLoss);
}
